// SimplifiedMamba2Block_56229711839693
// MI455X (gfx1250) — matrix operand order repaired and checked by reading, not compiled, not run on hardware
//
#include <hip/hip_runtime.h>

// ---------------------------------------------------------------------------
// Types for CDNA5 WMMA
// ---------------------------------------------------------------------------
typedef __attribute__((ext_vector_type(16))) __bf16 v16bf;
typedef __attribute__((ext_vector_type(8)))  float  v8f;

union FragB16 { v16bf v; unsigned int u[8]; };

#define AS3 __attribute__((address_space(3)))

// ---------------------------------------------------------------------------
// CDNA5 async global->LDS copy (ASYNCcnt path), via inline asm so we do not
// depend on the builtin's address-space-qualified pointer signature.
//   global_load_async_to_lds_b128 vdst(LDS byte offset), vaddr(64b), off
// ---------------------------------------------------------------------------
__device__ __forceinline__ unsigned int lds_byte_off(const void* p) {
#if defined(__gfx1250__)
    return (unsigned int)(unsigned long long)(AS3 const void*)p;   // ptrtoint(AS3)
#else
    (void)p; return 0;
#endif
}

__device__ __forceinline__ void async_copy_b128(unsigned int lds_off, const void* g) {
#if defined(__gfx1250__)
    asm volatile("global_load_async_to_lds_b128 %0, %1, off"
                 :: "v"(lds_off), "v"((unsigned long long)g)
                 : "memory");
#else
    (void)lds_off; (void)g;
#endif
}

template <int N>
__device__ __forceinline__ void wait_asynccnt() {
#if defined(__gfx1250__)
#if __has_builtin(__builtin_amdgcn_s_wait_asynccnt)
    __builtin_amdgcn_s_wait_asynccnt(N);
#else
    asm volatile("s_wait_asynccnt %0" ::"n"(N));
#endif
#endif
}

// ---------------------------------------------------------------------------
// bf16 helpers (round-to-nearest-even), bit-level so we control precision
// ---------------------------------------------------------------------------
__device__ __forceinline__ unsigned short f32_to_bf16(float f) {
    unsigned int u = __float_as_uint(f);
    unsigned int lsb = (u >> 16) & 1u;
    u += 0x7fffu + lsb;
    return (unsigned short)(u >> 16);
}
__device__ __forceinline__ float bf16_to_f32(unsigned short h) {
    return __uint_as_float(((unsigned int)h) << 16);
}

// ---------------------------------------------------------------------------
// Split a fp32 tensor into bf16 hi + bf16 lo (x ~= hi + lo, ~16 mantissa bits)
// ---------------------------------------------------------------------------
__global__ __launch_bounds__(256) void split_bf16_kernel(
    const float* __restrict__ src, unsigned short* __restrict__ hi,
    unsigned short* __restrict__ lo, int n)
{
    int i = blockIdx.x * 256 + threadIdx.x;
    if (i < n) {
        float f = src[i];
        unsigned short h = f32_to_bf16(f);
        float fh = bf16_to_f32(h);
        unsigned short l = f32_to_bf16(f - fh);
        hi[i] = h;
        lo[i] = l;
    }
}

// ---------------------------------------------------------------------------
// LayerNorm over D=1024 + bf16 hi/lo split. One 256-thread block per row.
// ---------------------------------------------------------------------------
__global__ __launch_bounds__(256) void ln_split_kernel(
    const float* __restrict__ x, const float* __restrict__ w,
    const float* __restrict__ b, unsigned short* __restrict__ xh,
    unsigned short* __restrict__ xl)
{
    const int D = 1024;
    __shared__ float s1[256];
    __shared__ float s2[256];
    int tid = threadIdx.x;
    size_t base = (size_t)blockIdx.x * D;

    float4 v = reinterpret_cast<const float4*>(x + base)[tid];
    float sum  = v.x + v.y + v.z + v.w;
    float sumq = v.x * v.x + v.y * v.y + v.z * v.z + v.w * v.w;
    s1[tid] = sum; s2[tid] = sumq;
    __syncthreads();
    for (int s = 128; s > 0; s >>= 1) {
        if (tid < s) { s1[tid] += s1[tid + s]; s2[tid] += s2[tid + s]; }
        __syncthreads();
    }
    float mu  = s1[0] * (1.0f / D);
    float var = s2[0] * (1.0f / D) - mu * mu;
    float rstd = rsqrtf(var + 1e-5f);

    float4 wv = reinterpret_cast<const float4*>(w)[tid];
    float4 bv = reinterpret_cast<const float4*>(b)[tid];
    float xn0 = (v.x - mu) * rstd * wv.x + bv.x;
    float xn1 = (v.y - mu) * rstd * wv.y + bv.y;
    float xn2 = (v.z - mu) * rstd * wv.z + bv.z;
    float xn3 = (v.w - mu) * rstd * wv.w + bv.w;

    unsigned short h0 = f32_to_bf16(xn0), h1 = f32_to_bf16(xn1);
    unsigned short h2 = f32_to_bf16(xn2), h3 = f32_to_bf16(xn3);
    unsigned short l0 = f32_to_bf16(xn0 - bf16_to_f32(h0));
    unsigned short l1 = f32_to_bf16(xn1 - bf16_to_f32(h1));
    unsigned short l2 = f32_to_bf16(xn2 - bf16_to_f32(h2));
    unsigned short l3 = f32_to_bf16(xn3 - bf16_to_f32(h3));

    unsigned int* xh32 = reinterpret_cast<unsigned int*>(xh);
    unsigned int* xl32 = reinterpret_cast<unsigned int*>(xl);
    size_t d0 = (base >> 1) + (size_t)tid * 2;
    xh32[d0]     = (unsigned int)h0 | ((unsigned int)h1 << 16);
    xh32[d0 + 1] = (unsigned int)h2 | ((unsigned int)h3 << 16);
    xl32[d0]     = (unsigned int)l0 | ((unsigned int)l1 << 16);
    xl32[d0 + 1] = (unsigned int)l2 | ((unsigned int)l3 << 16);
}

// ---------------------------------------------------------------------------
// GEMM  C[M,N] = A[M,K] * B[N,K]^T  with 2-term bf16 split:
//   C = Ah*Bh + Ah*Bl + Al*Bh   (3x v_wmma_f32_16x16x32_bf16 per tile)
// Block: 128 threads (4 waves). Macro tile 64(M) x 64(N).
// K staged in 32-chunks through double-buffered LDS via
// global_load_async_to_lds_b128 (ASYNCcnt software pipeline); 20-dword row
// pitch -> 16B-aligned rows and conflict-free strided fragment reads.
// mode 0: N==4096, epilogue splits into x_ssm (n<2048, fp32) and silu(gate).
// mode 1: plain fp32 store to out0[M,N].
// ---------------------------------------------------------------------------
__global__ __launch_bounds__(128) void gemm_bf16x3_kernel(
    const unsigned short* __restrict__ Ahi, const unsigned short* __restrict__ Alo,
    const unsigned short* __restrict__ Bhi, const unsigned short* __restrict__ Blo,
    int M, int N, int K, int mode,
    float* __restrict__ out0, float* __restrict__ out1)
{
    const int PITCH = 20;                       // dwords per 32-bf16 row (+pad)
    __shared__ __align__(16) unsigned int lsAh[2][64 * PITCH];
    __shared__ __align__(16) unsigned int lsAl[2][64 * PITCH];
    __shared__ __align__(16) unsigned int lsBh[2][64 * PITCH];
    __shared__ __align__(16) unsigned int lsBl[2][64 * PITCH];

    const int tid  = threadIdx.x;
    const int lane = tid & 31;
    const int wave = tid >> 5;
    const int half = lane >> 4;
    const int l15  = lane & 15;

    const int rowA0 = blockIdx.y * 64;   // M block
    const int rowB0 = blockIdx.x * 64;   // N block (B stored [N,K] row-major)

    const unsigned int* gAh = reinterpret_cast<const unsigned int*>(Ahi);
    const unsigned int* gAl = reinterpret_cast<const unsigned int*>(Alo);
    const unsigned int* gBh = reinterpret_cast<const unsigned int*>(Bhi);
    const unsigned int* gBl = reinterpret_cast<const unsigned int*>(Blo);
    const int kd2 = K >> 1;              // global row pitch in dwords

    // Stage one 32-wide K chunk (4 tiles: Ah/Al/Bh/Bl) into LDS buffer `buf`.
    // 8 x async b128 per thread => s_wait_asynccnt 8 retires the previous chunk
    // (async loads complete in order).
    auto stage = [&](int kc, int buf) {
#pragma unroll
        for (int i = 0; i < 2; ++i) {
            int slot = tid + 128 * i;            // 0..255
            int r  = slot >> 2;                  // tile row 0..63
            int c4 = slot & 3;                   // 16B chunk 0..3
            size_t ga = (size_t)(rowA0 + r) * kd2 + (size_t)(kc << 4) + c4 * 4;
            size_t gb = (size_t)(rowB0 + r) * kd2 + (size_t)(kc << 4) + c4 * 4;
            int li = r * PITCH + c4 * 4;
#if defined(__gfx1250__)
            async_copy_b128(lds_byte_off(&lsAh[buf][li]), gAh + ga);
            async_copy_b128(lds_byte_off(&lsAl[buf][li]), gAl + ga);
            async_copy_b128(lds_byte_off(&lsBh[buf][li]), gBh + gb);
            async_copy_b128(lds_byte_off(&lsBl[buf][li]), gBl + gb);
#else
            *reinterpret_cast<uint4*>(&lsAh[buf][li]) =
                *reinterpret_cast<const uint4*>(gAh + ga);
            *reinterpret_cast<uint4*>(&lsAl[buf][li]) =
                *reinterpret_cast<const uint4*>(gAl + ga);
            *reinterpret_cast<uint4*>(&lsBh[buf][li]) =
                *reinterpret_cast<const uint4*>(gBh + gb);
            *reinterpret_cast<uint4*>(&lsBl[buf][li]) =
                *reinterpret_cast<const uint4*>(gBl + gb);
#endif
        }
    };

    v8f acc[4];
    acc[0] = (v8f){}; acc[1] = (v8f){}; acc[2] = (v8f){}; acc[3] = (v8f){};

    const int nk = K >> 5;               // number of 32-wide K chunks
    stage(0, 0);                         // prologue fill

    for (int kc = 0; kc < nk; ++kc) {
        const int buf = kc & 1;
        const bool has_next = (kc + 1) < nk;
        if (has_next) stage(kc + 1, buf ^ 1);
        if (has_next) wait_asynccnt<8>(); else wait_asynccnt<0>();
        __syncthreads();                 // chunk kc visible to all waves

        // A fragment (16x32 bf16): VGPR j<4 -> K=2j(+8*half); j>=4 -> 16+2(j-4)(+8*half)
        FragB16 ah, al;
        int arow = wave * 16 + l15;
#pragma unroll
        for (int j = 0; j < 8; ++j) {
            int kd = (j < 4 ? j : 8 + (j - 4)) + 4 * half;   // dword in row
            ah.u[j] = lsAh[buf][arow * PITCH + kd];
            al.u[j] = lsAl[buf][arow * PITCH + kd];
        }
        // B fragments (32x16 bf16): lane n = l15, VGPR j<4 -> K=2j(+8*half); j>=4 -> 16+2(j-4)(+8*half)
        FragB16 bh[4], bl[4];
#pragma unroll
        for (int nt = 0; nt < 4; ++nt) {
            int brow = nt * 16 + l15;
#pragma unroll
            for (int j = 0; j < 8; ++j) {
                int kd = (j < 4 ? j : 8 + (j - 4)) + 4 * half;
                bh[nt].u[j] = lsBh[buf][brow * PITCH + kd];
                bl[nt].u[j] = lsBl[buf][brow * PITCH + kd];
            }
        }
#pragma unroll
        for (int nt = 0; nt < 4; ++nt) {
            acc[nt] = __builtin_amdgcn_wmma_f32_16x16x32_bf16(
                false, ah.v, false, bh[nt].v, (short)0, acc[nt], false, false);
            acc[nt] = __builtin_amdgcn_wmma_f32_16x16x32_bf16(
                false, ah.v, false, bl[nt].v, (short)0, acc[nt], false, false);
            acc[nt] = __builtin_amdgcn_wmma_f32_16x16x32_bf16(
                false, al.v, false, bh[nt].v, (short)0, acc[nt], false, false);
        }
        __syncthreads();                 // all waves done with buf before refill
    }

    // Epilogue: C layout — VGPR r: M = r + 8*half, N = l15
#pragma unroll
    for (int r = 0; r < 8; ++r) {
        int m = rowA0 + wave * 16 + half * 8 + r;
#pragma unroll
        for (int nt = 0; nt < 4; ++nt) {
            int n = rowB0 + nt * 16 + l15;
            float v = acc[nt][r];
            if (mode == 0) {
                if (n < 2048) {
                    out0[(size_t)m * 2048 + n] = v;                  // x_ssm
                } else {
                    float g = v / (1.0f + expf(-v));                 // silu(gate)
                    out1[(size_t)m * 2048 + (n - 2048)] = g;
                }
            } else {
                out0[(size_t)m * N + n] = v;
            }
        }
    }
}

// ---------------------------------------------------------------------------
// Bidirectional SSM scan. One lane per channel; 16 states in registers.
// grid = (d_inner/256, B, 2), block = 256.  A = -clip(exp(A_log),-10,-0.01),
// h = clip(h*A + x*Bm, -10, 10), y = <h, C>.  Coalesced per-step x loads,
// global_prefetch a few steps ahead.
// ---------------------------------------------------------------------------
__global__ __launch_bounds__(256) void ssm_scan_kernel(
    const float* __restrict__ x_ssm, const float* __restrict__ A_log,
    const float* __restrict__ Bm, const float* __restrict__ Cm,
    float* __restrict__ y_f, float* __restrict__ y_b, int L, int di)
{
    const int DS = 16;
    int c   = blockIdx.x * 256 + threadIdx.x;   // channel in d_inner
    int b   = blockIdx.y;
    int rev = blockIdx.z;                        // 0 = forward, 1 = backward
    float* __restrict__ yo = rev ? y_b : y_f;

    float a[DS], bm[DS], cm[DS], h[DS];
#pragma unroll
    for (int s = 0; s < DS; ++s) {
        float e = expf(A_log[(size_t)c * DS + s]);
        a[s]  = -fminf(fmaxf(e, -10.0f), -0.01f);    // faithful clip semantics
        bm[s] = Bm[(size_t)c * DS + s];
        cm[s] = Cm[(size_t)c * DS + s];
        h[s]  = 0.0f;
    }

    const size_t rowStride = (size_t)di;
    const size_t base = (size_t)b * L * di + c;
    for (int t = 0; t < L; ++t) {
        int tt = rev ? (L - 1 - t) : t;
        size_t off = base + (size_t)tt * rowStride;
        float x = x_ssm[off];
        // prefetch 8 steps ahead (global_prefetch_b8)
        int tp = rev ? (tt - 8) : (tt + 8);
        if (tp >= 0 && tp < L)
            __builtin_prefetch(&x_ssm[base + (size_t)tp * rowStride], 0, 1);
        float y = 0.0f;
#pragma unroll
        for (int s = 0; s < DS; ++s) {
            float hv = fmaf(h[s], a[s], x * bm[s]);
            hv = fminf(fmaxf(hv, -10.0f), 10.0f);
            h[s] = hv;
            y = fmaf(hv, cm[s], y);
        }
        yo[off] = y;
    }
}

// ---------------------------------------------------------------------------
// u = (y_f + y_b) * 0.5 * silu(gate)  -> bf16 hi/lo for GEMM2
// ---------------------------------------------------------------------------
__global__ __launch_bounds__(256) void combine_kernel(
    const float* __restrict__ yf, const float* __restrict__ yb,
    const float* __restrict__ g, unsigned short* __restrict__ uh,
    unsigned short* __restrict__ ul, int n)
{
    int i = blockIdx.x * 256 + threadIdx.x;
    if (i < n) {
        float u = (yf[i] + yb[i]) * 0.5f * g[i];
        unsigned short h = f32_to_bf16(u);
        unsigned short l = f32_to_bf16(u - bf16_to_f32(h));
        uh[i] = h;
        ul[i] = l;
    }
}

// ---------------------------------------------------------------------------
// Launch
// ---------------------------------------------------------------------------
extern "C" void kernel_launch(void* const* d_in, const int* in_sizes, int n_in,
                              void* d_out, int out_size, void* d_ws, size_t ws_size,
                              hipStream_t stream) {
    (void)in_sizes; (void)n_in; (void)out_size; (void)ws_size;
    const int B = 4, L = 2048, D = 1024, DI = 2048;
    const int M = B * L;                       // 8192 rows

    const float* x      = (const float*)d_in[0];
    const float* w1     = (const float*)d_in[1];  // [2*DI, D]
    const float* A_log  = (const float*)d_in[2];  // [DI, 16]
    const float* B_mat  = (const float*)d_in[3];
    const float* C_mat  = (const float*)d_in[4];
    const float* w2     = (const float*)d_in[5];  // [D, DI]
    const float* ln_w   = (const float*)d_in[6];
    const float* ln_b   = (const float*)d_in[7];
    float* out = (float*)d_out;

    char* p = (char*)d_ws;
    auto carve = [&](size_t bytes) {
        void* r = (void*)p;
        p += (bytes + 255) & ~(size_t)255;
        return r;
    };
    unsigned short* xnh = (unsigned short*)carve((size_t)M * D * 2);
    unsigned short* xnl = (unsigned short*)carve((size_t)M * D * 2);
    unsigned short* w1h = (unsigned short*)carve((size_t)2 * DI * D * 2);
    unsigned short* w1l = (unsigned short*)carve((size_t)2 * DI * D * 2);
    unsigned short* w2h = (unsigned short*)carve((size_t)D * DI * 2);
    unsigned short* w2l = (unsigned short*)carve((size_t)D * DI * 2);
    float* xssm = (float*)carve((size_t)M * DI * 4);
    float* gate = (float*)carve((size_t)M * DI * 4);
    float* yf   = (float*)carve((size_t)M * DI * 4);
    float* yb   = (float*)carve((size_t)M * DI * 4);
    unsigned short* uh = (unsigned short*)carve((size_t)M * DI * 2);
    unsigned short* ul = (unsigned short*)carve((size_t)M * DI * 2);

    // 1) weight splits
    {
        int n1 = 2 * DI * D;
        split_bf16_kernel<<<(n1 + 255) / 256, 256, 0, stream>>>(w1, w1h, w1l, n1);
        int n2 = D * DI;
        split_bf16_kernel<<<(n2 + 255) / 256, 256, 0, stream>>>(w2, w2h, w2l, n2);
    }
    // 2) LayerNorm + split
    ln_split_kernel<<<M, 256, 0, stream>>>(x, ln_w, ln_b, xnh, xnl);
    // 3) GEMM1: [M,D] x [4096,D]^T -> x_ssm + silu(gate)
    gemm_bf16x3_kernel<<<dim3((2 * DI) / 64, M / 64), 128, 0, stream>>>(
        xnh, xnl, w1h, w1l, M, 2 * DI, D, 0, xssm, gate);
    // 4) bidirectional scans
    ssm_scan_kernel<<<dim3(DI / 256, B, 2), 256, 0, stream>>>(
        xssm, A_log, B_mat, C_mat, yf, yb, L, DI);
    // 5) combine + split to bf16
    {
        int n = M * DI;
        combine_kernel<<<(n + 255) / 256, 256, 0, stream>>>(yf, yb, gate, uh, ul, n);
    }
    // 6) GEMM2: [M,DI] x [D,DI]^T -> out fp32
    gemm_bf16x3_kernel<<<dim3(D / 64, M / 64), 128, 0, stream>>>(
        uh, ul, w2h, w2l, M, D, DI, 1, out, nullptr);
}
